// CochlearModel_30580167147903
// MI455X (gfx1250) — matrix operand order repaired and checked by reading, not compiled, not run on hardware
//
#include <hip/hip_runtime.h>
#include <hip/hip_bf16.h>

// ---------------- config mirrored from the reference ----------------
#define NCH   64
#define TLEN  48000
#define ALPHA 0.04f        // DT/(DT+tau) = (1/48000)/((1/48000)+1/2000) = 1/25 exactly
#define TH0   0.1f
#define AR    0.01f

typedef __attribute__((ext_vector_type(16))) _Float16 v16h;
typedef __attribute__((ext_vector_type(8)))  _Float16 v8h;
typedef __attribute__((ext_vector_type(8)))  float    v8f;

union F16x16 { _Float16 h[16]; unsigned u[8]; v16h v; };

// CDNA5 async global->LDS path (ASYNCcnt-tracked), guarded so either
// toolchain still compiles; fallback is register-staged double buffering.
#if defined(__AMDGCN__) &&                                             \
    __has_builtin(__builtin_amdgcn_global_load_async_to_lds_b128) &&   \
    __has_builtin(__builtin_amdgcn_s_wait_asynccnt)
#define USE_ASYNC_LDS 1
#else
#define USE_ASYNC_LDS 0
#endif

#if USE_ASYNC_LDS
typedef int v4i __attribute__((vector_size(16)));   // matches builtin prototype
typedef __attribute__((address_space(1))) v4i* glb_b128_t;
typedef __attribute__((address_space(3))) v4i* lds_b128_t;
__device__ __forceinline__ void async_g2l_b128(const void* g, void* l) {
    // generic LDS pointer: low 32 bits are the LDS byte offset (ISA 10.2)
    __builtin_amdgcn_global_load_async_to_lds_b128(
        (glb_b128_t)(unsigned long long)g,
        (lds_b128_t)(unsigned)(unsigned long long)l, 0, 0);
}
#endif

// ---------------------------------------------------------------
// Stage 0: f32 -> f16 conversion + padding.
//   xpad[b][i] : length XPLEN = Lp + T, first L-1 zeros, then audio, zero tail
//   Kf[c][l]   : kernels, K-dim zero-padded to Lp (multiple of 64)
// ---------------------------------------------------------------
__global__ void prep_kernel(const float* __restrict__ audio,
                            const float* __restrict__ kern,
                            _Float16* __restrict__ xpad,
                            _Float16* __restrict__ Kf,
                            int L, int Lp, int XPLEN) {
    const int n_x   = 4 * XPLEN;
    const int n_k   = NCH * Lp;
    const int total = n_x + n_k;
    for (int idx = blockIdx.x * blockDim.x + threadIdx.x; idx < total;
         idx += gridDim.x * blockDim.x) {
        if (idx < n_x) {
            int b = idx / XPLEN;
            int i = idx - b * XPLEN;
            int j = i - (L - 1);
            float v = (j >= 0 && j < TLEN) ? audio[b * TLEN + j] : 0.0f;
            xpad[idx] = (_Float16)v;
        } else {
            int k  = idx - n_x;
            int ch = k / Lp;
            int l  = k - ch * Lp;
            Kf[k] = (_Float16)((l < L) ? kern[ch * L + l] : 0.0f);
        }
    }
}

// ---------------------------------------------------------------
// Stage 1: gammatone filterbank as implicit GEMM with WMMA.
//   D[c, t] = sum_l K[c,l] * xpad[b, t+l]  (M=64, N=T per batch, K=Lp)
// Block = 256 threads (8 waves) per (b, 128-t) tile.
// Wave w: M-tile = w&3 (16 channels), N-group = w>>2 (4 N-tiles of 16 t).
// K-chunk = 64 taps, double-buffered LDS, one barrier per chunk,
// next chunk streamed (async-to-LDS or via regs) under 8 WMMAs.
// ---------------------------------------------------------------
__device__ __forceinline__ void compute_chunk(const _Float16* A,
                                              const _Float16* X,
                                              int row, int kh, int g, int ln,
                                              v8f acc[4]) {
#pragma unroll
    for (int kk = 0; kk < 64; kk += 32) {
        // A fragment: contiguous k order, lane half h: u[j] holds k = 16h + 2j, +1
        F16x16 a;
#pragma unroll
        for (int j = 0; j < 8; ++j) {
            const int kb = (kh << 1) + (j << 1);
            a.u[j] = *(const unsigned*)&A[row * 64 + kk + kb];
        }
        // B fragments: contiguous k order (same as A) — lane n = ln&15,
        // VGPR j holds rows K = (ln>>4)*16 + 2j, +1
#pragma unroll
        for (int n = 0; n < 4; ++n) {
            F16x16 bf;
            const int base = kk + g * 64 + n * 16 + (ln & 15) + ((ln >> 4) << 4);
#pragma unroll
            for (int j = 0; j < 8; ++j) {
                bf.h[2 * j]     = X[base + 2 * j];
                bf.h[2 * j + 1] = X[base + 2 * j + 1];
            }
            acc[n] = __builtin_amdgcn_wmma_f32_16x16x32_f16(
                false, a.v, false, bf.v, (short)0, acc[n], false, false);
        }
    }
}

__global__ __launch_bounds__(256)
void fb_wmma_kernel(const _Float16* __restrict__ xpad,
                    const _Float16* __restrict__ Kf,
                    float* __restrict__ rect,
                    int Lp, int XPLEN) {
    __shared__ _Float16 ldsA[2][NCH * 64];   // 2 x 8KB: 64 ch x 64 taps
    __shared__ _Float16 ldsX[2][192];        // 2 x 384B: 128 t + 63 halo + pad

    const int tid  = threadIdx.x;
    const int b    = blockIdx.y;
    const int t0   = blockIdx.x * 128;
    const int wave = tid >> 5;
    const int ln   = tid & 31;
    const int mt   = wave & 3;               // M-tile (channels 16*mt..)
    const int g    = wave >> 2;              // N-group (64 t each)
    const int row  = mt * 16 + (ln & 15);
    const int kh   = (ln >> 4) << 3;

    const _Float16* xb = xpad + (size_t)b * XPLEN + t0;

    // staging assignment: 256 threads x 32B = the full 8KB A chunk
    const int sch   = tid >> 2;              // 0..63  channel row
    const int scol  = (tid & 3) << 4;        // 0,16,32,48 (16 halfs each)

    const int nk = Lp >> 6;
    v8f acc[4] = {};

#if USE_ASYNC_LDS
    // ---- prologue: async-stage chunk 0 into buffer 0 ----
    {
#pragma unroll
        for (int q = 0; q < 2; ++q)
            async_g2l_b128(&Kf[(size_t)sch * Lp + scol + q * 8],
                           &ldsA[0][sch * 64 + scol + q * 8]);
        if (tid < 24)
            async_g2l_b128(&xb[tid * 8], &ldsX[0][tid * 8]);
    }
    __builtin_amdgcn_s_wait_asynccnt(0);
    __syncthreads();

    for (int k = 0; k < nk; ++k) {
        const int cur = k & 1;
        const int k0n = (k + 1) << 6;
        if (k + 1 < nk) {  // async-stream next chunk into the other buffer
#pragma unroll
            for (int q = 0; q < 2; ++q)
                async_g2l_b128(&Kf[(size_t)sch * Lp + k0n + scol + q * 8],
                               &ldsA[cur ^ 1][sch * 64 + scol + q * 8]);
            if (tid < 24)
                async_g2l_b128(&xb[k0n + tid * 8], &ldsX[cur ^ 1][tid * 8]);
        }
        compute_chunk(ldsA[cur], ldsX[cur], row, kh, g, ln, acc);
        __builtin_amdgcn_s_wait_asynccnt(0);
        __syncthreads();
    }
#else
    // ---- fallback: register-staged double buffering ----
    v8h  ar[4];
    _Float16 xr = (_Float16)0.0f;
#pragma unroll
    for (int q = 0; q < 2; ++q) {
        ar[q] = *(const v8h*)&Kf[(size_t)sch * Lp + scol + q * 8];
        *(v8h*)&ldsA[0][sch * 64 + scol + q * 8] = ar[q];
    }
    if (tid < 192) ldsX[0][tid] = xb[tid];

    for (int k = 0; k < nk; ++k) {
        const int cur = k & 1;
        const int k0n = (k + 1) << 6;
        __syncthreads();
        if (k + 1 < nk) {
#pragma unroll
            for (int q = 0; q < 2; ++q)
                ar[q] = *(const v8h*)&Kf[(size_t)sch * Lp + k0n + scol + q * 8];
            if (tid < 192) xr = xb[k0n + tid];
            if (k + 2 < nk)
                __builtin_prefetch(&Kf[(size_t)sch * Lp + k0n + 64 + scol], 0, 1);
        }
        compute_chunk(ldsA[cur], ldsX[cur], row, kh, g, ln, acc);
        if (k + 1 < nk) {
#pragma unroll
            for (int q = 0; q < 2; ++q)
                *(v8h*)&ldsA[cur ^ 1][sch * 64 + scol + q * 8] = ar[q];
            if (tid < 192) ldsX[cur ^ 1][tid] = xr;
        }
    }
#endif

    // D layout: VGPR r -> M = r + (ln>=16 ? 8 : 0), N = ln&15
    const int crow = mt * 16 + ((ln >> 4) << 3);
    const int tcol = ln & 15;
#pragma unroll
    for (int n = 0; n < 4; ++n) {
        const int t = t0 + g * 64 + n * 16 + tcol;
        float* dst  = rect + (size_t)t * 256 + b * 64 + crow;
#pragma unroll
        for (int r = 0; r < 8; ++r)
            dst[r] = fmaxf(acc[n][r], 0.0f);   // ReLU fused into the store
    }
}

// ---------------------------------------------------------------
// Stage 2: sequential hair-cell scan. Channels independent -> 64 blocks
// of one wave32 each; lanes 0..3 carry the 4 batch states, all 32 lanes
// prefetch 32 timesteps into LDS per chunk (rect lives in the 192MB L2).
// Both ears reuse the same rect; carry persists across ears.
// ---------------------------------------------------------------
__global__ __launch_bounds__(32)
void scan_kernel(const float* __restrict__ rect, float* __restrict__ out) {
    const int c    = blockIdx.x;
    const int lane = threadIdx.x;
    __shared__ float sm[128];

    float s = 0.0f, tim = 0.0f, thr = TH0;

    float* spike = out;                            // [B][T][2C]
    float* resp  = out + (size_t)4 * TLEN * 128;   // [B][2][T][C]

    for (int e = 0; e < 2; ++e) {
        for (int t0 = 0; t0 < TLEN; t0 += 32) {
#pragma unroll
            for (int i = 0; i < 4; ++i) {
                const int ee = lane + 32 * i;      // ee = t'*4 + b
                const int tt = t0 + (ee >> 2);
                const int bb = ee & 3;
                sm[ee] = rect[(size_t)tt * 256 + bb * 64 + c];
            }
            __syncthreads();
            if (lane < 4) {
                const int b = lane;
#pragma unroll 4
                for (int q = 0; q < 32; ++q) {
                    const float xt = sm[q * 4 + b];
                    s = ALPHA * xt + (1.0f - ALPHA) * s;     // hair-cell LPF
                    const bool  refr = tim > 0.0f;
                    const float spk  = (s > thr && !refr) ? 1.0f : 0.0f;
                    float sum = spk;                          // batch reduce
                    sum += __shfl_xor(sum, 1, 4);
                    sum += __shfl_xor(sum, 2, 4);
                    if (sum > 0.0f) tim = 1.0f;               // REFRAC
                    tim = fmaxf(tim - 1.0f, 0.0f);
                    thr = thr + AR * (sum * 0.25f) + AR * (TH0 - thr);
                    const int t = t0 + q;
                    spike[((size_t)b * TLEN + t) * 128 + e * 64 + c] = spk;
                    resp[(((size_t)b * 2 + e) * TLEN + t) * 64 + c]  = s;
                }
            }
            __syncthreads();
        }
    }
}

// ---------------------------------------------------------------
extern "C" void kernel_launch(void* const* d_in, const int* in_sizes, int n_in,
                              void* d_out, int out_size, void* d_ws, size_t ws_size,
                              hipStream_t stream) {
    const float* audio = (const float*)d_in[0];   // [4, 48000] f32
    const float* kern  = (const float*)d_in[1];   // [64, L]    f32

    const int L     = in_sizes[1] / NCH;
    const int Lp    = (L + 63) & ~63;             // K padded to multiple of 64
    const int XPLEN = Lp + TLEN;

    // workspace layout: rect f32 [T][B][C] (49.2MB) | xpad f16 | Kf f16
    float*    rect = (float*)d_ws;
    _Float16* xpad = (_Float16*)(rect + (size_t)TLEN * 256);
    _Float16* Kf   = xpad + (size_t)4 * XPLEN;

    const int total = 4 * XPLEN + NCH * Lp;
    prep_kernel<<<(total + 255) / 256, 256, 0, stream>>>(audio, kern, xpad, Kf,
                                                         L, Lp, XPLEN);
    fb_wmma_kernel<<<dim3(TLEN / 128, 4), 256, 0, stream>>>(xpad, Kf, rect,
                                                            Lp, XPLEN);
    scan_kernel<<<NCH, 32, 0, stream>>>(rect, (float*)d_out);
}
